// EncodeProcessDecodeHistory_27496380629764
// MI455X (gfx1250) — compile-verified, matrix operand order checked by reading, not run on hardware
//
#include <hip/hip_runtime.h>
#include <hip/hip_bf16.h>

typedef __attribute__((ext_vector_type(16))) _Float16 v16h;
typedef __attribute__((ext_vector_type(8)))  float    v8f;

#define H 128
#define TM 128   // rows per block
#define KC 32    // K chunk

// ---------------------------------------------------------------------------
// CDNA5 16-bit WMMA layouts (wave32):
// A 16x32 f16: lane L -> M = L&15, half = L>>4; per lane the 16 f16 cover cols
//   [8h..8h+7] and [16+8h..16+8h+7] of the 32-wide K chunk, natural order.
// B 32x16 f16: symmetric with N in place of M; W pre-packed into global
//   fragment order [kc][tile][lane][16] so a b-frag is one contiguous 32B load.
// C/D 16x16 f32: reg r -> M = r + 8*(L>>4), N = L&15.
// ---------------------------------------------------------------------------

// Pack W [K,128] f32 (rows >= Kw treated as zero) into fragment-ordered f16.
__global__ void pack_w_k(const float* __restrict__ W, _Float16* __restrict__ Wf,
                         int K, int Kw)
{
    const size_t idx = (size_t)blockIdx.x * 256 + threadIdx.x;
    if (idx >= (size_t)K * H) return;
    const int kc   = (int)(idx >> 12);          // /4096
    const int rem  = (int)(idx & 4095);
    const int tile = rem >> 9;                  // /512
    const int rem2 = rem & 511;
    const int lane = rem2 >> 4;
    const int i    = rem2 & 15;
    const int hl   = lane >> 4;
    const int k    = kc * KC + (i < 8 ? 8 * hl + i : 16 + 8 * hl + (i - 8));
    const int n    = tile * 16 + (lane & 15);
    Wf[idx] = (k < Kw) ? (_Float16)W[(size_t)k * H + n] : (_Float16)0.0f;
}

// ---------------------------------------------------------------------------
// LDS-free tiled GEMM: C[M,128] = act(A[M,K] @ W + bias), W pre-packed (Wf).
// mode 0: row m of A at A + m*lda
// mode 1: cols 0-127 = Xh[rIdx[m]], 128-255 = Xh[cIdx[m]], 256-383 = P2[m]
// mode 2: cols 0-127 = Xh[m], 128-255 = P2[m]
// A fragments are software-pipelined one K-chunk ahead so the (possibly
// random-gather) HBM latency overlaps the 8 WMMAs of the current chunk.
// ---------------------------------------------------------------------------
__global__ __launch_bounds__(256) void gemm_tile_k(
    const float* __restrict__ A, int lda,
    const float* __restrict__ Xh, const float* __restrict__ P2,
    const int* __restrict__ rIdx, const int* __restrict__ cIdx, int mode,
    const _Float16* __restrict__ Wf, const float* __restrict__ bias,
    float* __restrict__ C, int M, int K, int relu)
{
    const int t    = threadIdx.x;
    const int wv   = t >> 5;
    const int lane = t & 31;
    const int half = lane >> 4;
    const int m    = blockIdx.x * TM + wv * 16 + (lane & 15);
    const bool valid = (m < M);

    // per-thread source base pointers (row m)
    const float* b0 = nullptr;
    const float* b1 = nullptr;
    const float* b2 = nullptr;
    if (valid) {
        if (mode == 0) {
            b0 = A + (size_t)m * lda;
        } else if (mode == 1) {
            b0 = Xh + (size_t)rIdx[m] * H;
            b1 = Xh + (size_t)cIdx[m] * H;
            b2 = P2 + (size_t)m * H;
        } else {
            b0 = Xh + (size_t)m * H;
            b1 = P2 + (size_t)m * H;
        }
    }
    auto srcp = [&](int kc) -> const float* {
        if (mode == 0) return b0 + kc * KC;
        if (mode == 1) {
            const int sub = kc >> 2;
            return (sub == 0 ? b0 : (sub == 1 ? b1 : b2)) + (kc & 3) * KC;
        }
        return ((kc >> 2) == 0 ? b0 : b1) + (kc & 3) * KC;
    };

    v8f acc[8];
#pragma unroll
    for (int n = 0; n < 8; ++n) acc[n] = (v8f){0.f,0.f,0.f,0.f,0.f,0.f,0.f,0.f};

    const float4 fz = {0.f, 0.f, 0.f, 0.f};
    float4 q0 = fz, q1 = fz, q2 = fz, q3 = fz;
    if (valid) {
        const float* s = srcp(0);
        q0 = *(const float4*)(s + 8 * half);
        q1 = *(const float4*)(s + 8 * half + 4);
        q2 = *(const float4*)(s + 16 + 8 * half);
        q3 = *(const float4*)(s + 16 + 8 * half + 4);
    }

    const int nChunks = K >> 5;
    for (int kc = 0; kc < nChunks; ++kc) {
        // ---- convert prefetched A data to f16 fragment ----
        v16h a;
        a[0]  = (_Float16)q0.x; a[1]  = (_Float16)q0.y;
        a[2]  = (_Float16)q0.z; a[3]  = (_Float16)q0.w;
        a[4]  = (_Float16)q1.x; a[5]  = (_Float16)q1.y;
        a[6]  = (_Float16)q1.z; a[7]  = (_Float16)q1.w;
        a[8]  = (_Float16)q2.x; a[9]  = (_Float16)q2.y;
        a[10] = (_Float16)q2.z; a[11] = (_Float16)q2.w;
        a[12] = (_Float16)q3.x; a[13] = (_Float16)q3.y;
        a[14] = (_Float16)q3.z; a[15] = (_Float16)q3.w;

        // ---- issue next chunk's A gather now; consumed next iteration ----
        if (kc + 1 < nChunks && valid) {
            const float* s = srcp(kc + 1);
            q0 = *(const float4*)(s + 8 * half);
            q1 = *(const float4*)(s + 8 * half + 4);
            q2 = *(const float4*)(s + 16 + 8 * half);
            q3 = *(const float4*)(s + 16 + 8 * half + 4);
        }

        // ---- B fragments (WGP$-resident packed weights) + WMMAs ----
        const _Float16* wp = Wf + (size_t)kc * 4096 + lane * 16;
        if (kc + 1 < nChunks)
            __builtin_prefetch(wp + 4096, 0, 0);   // global_prefetch_b8

        v16h bg0[4], bg1[4];
#pragma unroll
        for (int j = 0; j < 4; ++j) bg0[j] = *(const v16h*)(wp + j * 512);
#pragma unroll
        for (int j = 0; j < 4; ++j) bg1[j] = *(const v16h*)(wp + (4 + j) * 512);
#pragma unroll
        for (int j = 0; j < 4; ++j)
            acc[j] = __builtin_amdgcn_wmma_f32_16x16x32_f16(
                false, a, false, bg0[j], (short)0, acc[j], false, false);
#pragma unroll
        for (int j = 0; j < 4; ++j)
            acc[4 + j] = __builtin_amdgcn_wmma_f32_16x16x32_f16(
                false, a, false, bg1[j], (short)0, acc[4 + j], false, false);
    }

    // ---- epilogue: bias + optional ReLU ----
    const int nl = lane & 15;
    const int mb = blockIdx.x * TM + wv * 16;
#pragma unroll
    for (int n = 0; n < 8; ++n) {
        const float bv = bias[n * 16 + nl];
#pragma unroll
        for (int r = 0; r < 8; ++r) {
            const int mm = mb + r + 8 * half;
            if (mm < M) {
                float v = acc[n][r] + bv;
                if (relu) v = fmaxf(v, 0.0f);
                C[(size_t)mm * H + n * 16 + nl] = v;
            }
        }
    }
}

// ---------------------------------------------------------------------------
// LayerNorm over last dim (128), wave32 per row, optional post-LN residual.
// ---------------------------------------------------------------------------
__global__ __launch_bounds__(256) void layernorm_k(
    const float* __restrict__ X, const float* __restrict__ g,
    const float* __restrict__ b, const float* __restrict__ res,
    float* __restrict__ Y, int M)
{
    const int row  = blockIdx.x * 8 + (threadIdx.x >> 5);
    const int lane = threadIdx.x & 31;
    if (row >= M) return;
    const float4 v = ((const float4*)(X + (size_t)row * H))[lane];
    float s = v.x + v.y + v.z + v.w;
#pragma unroll
    for (int off = 16; off; off >>= 1) s += __shfl_xor(s, off, 32);
    const float mean = s * (1.0f / H);
    const float d0 = v.x - mean, d1 = v.y - mean, d2 = v.z - mean, d3 = v.w - mean;
    float q = d0 * d0 + d1 * d1 + d2 * d2 + d3 * d3;
#pragma unroll
    for (int off = 16; off; off >>= 1) q += __shfl_xor(q, off, 32);
    const float inv = rsqrtf(q * (1.0f / H) + 1e-5f);
    const float4 gg = ((const float4*)g)[lane];
    const float4 bb = ((const float4*)b)[lane];
    float4 o;
    o.x = d0 * inv * gg.x + bb.x;
    o.y = d1 * inv * gg.y + bb.y;
    o.z = d2 * inv * gg.z + bb.z;
    o.w = d3 * inv * gg.w + bb.w;
    if (res) {
        const float4 rr = ((const float4*)(res + (size_t)row * H))[lane];
        o.x += rr.x; o.y += rr.y; o.z += rr.z; o.w += rr.w;
    }
    ((float4*)(Y + (size_t)row * H))[lane] = o;
}

// ---------------------------------------------------------------------------
// Feature builders, scatter-add, misc elementwise.
// ---------------------------------------------------------------------------
__global__ void node_feat_k(const float* __restrict__ phi, const float* __restrict__ pphi,
                            const float* __restrict__ sphi, const float* __restrict__ srate,
                            const float* __restrict__ sratep, const float* __restrict__ ntype,
                            float* __restrict__ xf, int N)
{
    const int i = blockIdx.x * 256 + threadIdx.x;
    if (i >= N) return;
    float* o = xf + (size_t)i * 32;
    o[0] = phi[i];
    o[1] = phi[i] - pphi[i];
    o[2] = sphi[i];
    o[3] = srate[i];
    o[4] = sratep[i];
    o[5] = ntype[(size_t)i * 4 + 0];
    o[6] = ntype[(size_t)i * 4 + 1];
    o[7] = ntype[(size_t)i * 4 + 2];
    o[8] = ntype[(size_t)i * 4 + 3];
#pragma unroll
    for (int j = 9; j < 32; ++j) o[j] = 0.0f;
}

__global__ void edge_feat_k(const float* __restrict__ mesh, const float* __restrict__ world,
                            const float* __restrict__ phi, const int* __restrict__ erow,
                            const int* __restrict__ ecol, float* __restrict__ ef, int E)
{
    const int e = blockIdx.x * 256 + threadIdx.x;
    if (e >= E) return;
    const int r = erow[e], c = ecol[e];
    const float rpx = mesh[2 * r] - mesh[2 * c];
    const float rpy = mesh[2 * r + 1] - mesh[2 * c + 1];
    const float rwx = world[2 * r] - world[2 * c];
    const float rwy = world[2 * r + 1] - world[2 * c + 1];
    float* o = ef + (size_t)e * 32;
    o[0] = rpx; o[1] = rpy; o[2] = sqrtf(rpx * rpx + rpy * rpy);
    o[3] = rwx; o[4] = rwy; o[5] = sqrtf(rwx * rwx + rwy * rwy);
    o[6] = phi[r] - phi[c];
#pragma unroll
    for (int j = 7; j < 32; ++j) o[j] = 0.0f;
}

__global__ void zero_k(float* __restrict__ p, size_t n) {
    const size_t i = (size_t)blockIdx.x * 256 + threadIdx.x;
    if (i < n) p[i] = 0.0f;
}

__global__ void scatter_add_k(const float* __restrict__ newe, const int* __restrict__ ecol,
                              float* __restrict__ aggr, int E)
{
    const size_t i = (size_t)blockIdx.x * 256 + threadIdx.x;
    if (i >= (size_t)E * H) return;
    const int e = (int)(i >> 7);
    const int j = (int)(i & (H - 1));
    atomicAdd(&aggr[(size_t)ecol[e] * H + j], newe[i]);
}

__global__ void add_inplace_k(float* __restrict__ dst, const float* __restrict__ src, size_t n) {
    const size_t i = (size_t)blockIdx.x * 256 + threadIdx.x;
    if (i < n) dst[i] += src[i];
}

// out[i*3 + outOff + c] = h[i,:] . w2[:,c] + b2[c],  w2: [128, nout] row-major
__global__ void dec_out_k(const float* __restrict__ h, const float* __restrict__ w2,
                          const float* __restrict__ b2, float* __restrict__ out,
                          int N, int nout, int outOff)
{
    const size_t idx = (size_t)blockIdx.x * 256 + threadIdx.x;
    if (idx >= (size_t)N * nout) return;
    const int i = (int)(idx / nout);
    const int c = (int)(idx % nout);
    const float* hr = h + (size_t)i * H;
    float s = b2[c];
#pragma unroll 4
    for (int k = 0; k < H; ++k) s += hr[k] * w2[(size_t)k * nout + c];
    out[(size_t)i * 3 + outOff + c] = s;
}

// ---------------------------------------------------------------------------
// Host orchestration
// ---------------------------------------------------------------------------
static inline void launch_gemm(hipStream_t st, const float* A, int lda,
                               const float* Xh, const float* P2,
                               const int* rIdx, const int* cIdx, int mode,
                               const _Float16* Wf, const float* bias,
                               float* C, int M, int K, int relu)
{
    dim3 grid((M + TM - 1) / TM);
    gemm_tile_k<<<grid, 256, 0, st>>>(A, lda, Xh, P2, rIdx, cIdx, mode,
                                      Wf, bias, C, M, K, relu);
}

extern "C" void kernel_launch(void* const* d_in, const int* in_sizes, int n_in,
                              void* d_out, int out_size, void* d_ws, size_t ws_size,
                              hipStream_t stream)
{
    (void)n_in; (void)out_size; (void)ws_size;
    const float* world  = (const float*)d_in[0];
    const float* mesh   = (const float*)d_in[1];
    const float* phi    = (const float*)d_in[3];
    const float* pphi   = (const float*)d_in[4];
    const float* sphi   = (const float*)d_in[5];
    const float* srate  = (const float*)d_in[6];
    const float* sratep = (const float*)d_in[7];
    const float* ntype  = (const float*)d_in[8];
    const int*   eidx   = (const int*)d_in[9];
    const int N = in_sizes[3];
    const int E = in_sizes[9] / 2;
    const int* erow = eidx;
    const int* ecol = eidx + E;

#define PF(i) ((const float*)d_in[i])
    // params, JAX pytree (alphabetical) order
    const float *dp_b1 = PF(10), *dp_b2 = PF(11), *dp_w1 = PF(12), *dp_w2 = PF(13);
    const float *dw_b1 = PF(14), *dw_b2 = PF(15), *dw_w1 = PF(16), *dw_w2 = PF(17);
    const float *ee_b1 = PF(18), *ee_b2 = PF(19), *ee_lnb = PF(20), *ee_lng = PF(21),
                *ee_w1 = PF(22), *ee_w2 = PF(23);
    const float *ne_b1 = PF(24), *ne_b2 = PF(25), *ne_lnb = PF(26), *ne_lng = PF(27),
                *ne_w1 = PF(28), *ne_w2 = PF(29);
    // proc[s]: base 30 + 12*s : edge_mlp{b1,b2,ln_b,ln_g,w1,w2}, node_mlp{...}

    // ---- workspace carve ----
    float* w = (float*)d_ws;
    size_t off = 0;
    auto carve = [&](size_t n) { float* p = w + off; off += n; return p; };
    float* xf    = carve((size_t)N * 32);
    float* ef    = carve((size_t)E * 32);
    float* xhA   = carve((size_t)N * H);
    float* xhB   = carve((size_t)N * H);
    float* eh    = carve((size_t)E * H);
    float* tmpE1 = carve((size_t)E * H);
    float* tmpE2 = carve((size_t)E * H);
    float* tmpN1 = carve((size_t)N * H);
    float* tmpN2 = carve((size_t)N * H);
    float* aggr  = carve((size_t)N * H);
    off = (off + 7) & ~(size_t)7;   // 32B align for f16 fragment buffers

    // ---- pack all GEMM weights into fragment-ordered f16 ----
    auto pack = [&](const float* W, int K, int Kw) -> const _Float16* {
        _Float16* Wf = (_Float16*)carve(((size_t)K * H + 1) / 2);
        const int nthr = (int)(((size_t)K * H + 255) / 256);
        pack_w_k<<<nthr, 256, 0, stream>>>(W, Wf, K, Kw);
        return Wf;
    };
    const _Float16* ne_w1f = pack(ne_w1, 32, 9);
    const _Float16* ne_w2f = pack(ne_w2, H, H);
    const _Float16* ee_w1f = pack(ee_w1, 32, 7);
    const _Float16* ee_w2f = pack(ee_w2, H, H);
    const _Float16* em_w1f[3]; const _Float16* em_w2f[3];
    const _Float16* nm_w1f[3]; const _Float16* nm_w2f[3];
    for (int s = 0; s < 3; ++s) {
        const int base = 30 + 12 * s;
        em_w1f[s] = pack(PF(base + 4), 3 * H, 3 * H);
        em_w2f[s] = pack(PF(base + 5), H, H);
        nm_w1f[s] = pack(PF(base + 10), 2 * H, 2 * H);
        nm_w2f[s] = pack(PF(base + 11), H, H);
    }
    const _Float16* dw_w1f = pack(dw_w1, H, H);
    const _Float16* dp_w1f = pack(dp_w1, H, H);

    const int nb  = (N + 255) / 256;
    const int ebk = (E + 255) / 256;

    // ---- features ----
    node_feat_k<<<nb, 256, 0, stream>>>(phi, pphi, sphi, srate, sratep, ntype, xf, N);
    edge_feat_k<<<ebk, 256, 0, stream>>>(mesh, world, phi, erow, ecol, ef, E);

    // ---- node encoder: Lin-ReLU-Lin-LN ----
    launch_gemm(stream, xf, 32, nullptr, nullptr, nullptr, nullptr, 0, ne_w1f, ne_b1, tmpN1, N, 32, 1);
    launch_gemm(stream, tmpN1, H, nullptr, nullptr, nullptr, nullptr, 0, ne_w2f, ne_b2, tmpN2, N, H, 0);
    layernorm_k<<<(N + 7) / 8, 256, 0, stream>>>(tmpN2, ne_lng, ne_lnb, nullptr, xhA, N);

    // ---- edge encoder ----
    launch_gemm(stream, ef, 32, nullptr, nullptr, nullptr, nullptr, 0, ee_w1f, ee_b1, tmpE1, E, 32, 1);
    launch_gemm(stream, tmpE1, H, nullptr, nullptr, nullptr, nullptr, 0, ee_w2f, ee_b2, tmpE2, E, H, 0);
    layernorm_k<<<(E + 7) / 8, 256, 0, stream>>>(tmpE2, ee_lng, ee_lnb, nullptr, eh, E);

    // ---- processor steps ----
    float* xh_cur = xhA;
    float* xh_nxt = xhB;
    for (int s = 0; s < 3; ++s) {
        const int base = 30 + 12 * s;
        const float *em_b1 = PF(base + 0), *em_b2 = PF(base + 1), *em_lnb = PF(base + 2),
                    *em_lng = PF(base + 3);
        const float *nm_b1 = PF(base + 6), *nm_b2 = PF(base + 7), *nm_lnb = PF(base + 8),
                    *nm_lng = PF(base + 9);

        // edge MLP on [x_h[row] | x_h[col] | e_h]  (gathered per-thread, no big buffer)
        launch_gemm(stream, nullptr, 0, xh_cur, eh, erow, ecol, 1, em_w1f[s], em_b1, tmpE1, E, 3 * H, 1);
        launch_gemm(stream, tmpE1, H, nullptr, nullptr, nullptr, nullptr, 0, em_w2f[s], em_b2, tmpE2, E, H, 1);
        layernorm_k<<<(E + 7) / 8, 256, 0, stream>>>(tmpE2, em_lng, em_lnb, nullptr, tmpE2, E); // new_e

        // aggregate new_e by receiver
        zero_k<<<(int)(((size_t)N * H + 255) / 256), 256, 0, stream>>>(aggr, (size_t)N * H);
        scatter_add_k<<<(int)(((size_t)E * H + 255) / 256), 256, 0, stream>>>(tmpE2, ecol, aggr, E);

        // node MLP on [x_h | aggr], residual after LN
        launch_gemm(stream, nullptr, 0, xh_cur, aggr, nullptr, nullptr, 2, nm_w1f[s], nm_b1, tmpN1, N, 2 * H, 1);
        launch_gemm(stream, tmpN1, H, nullptr, nullptr, nullptr, nullptr, 0, nm_w2f[s], nm_b2, tmpN2, N, H, 1);
        layernorm_k<<<(N + 7) / 8, 256, 0, stream>>>(tmpN2, nm_lng, nm_lnb, xh_cur, xh_nxt, N);
        float* t = xh_cur; xh_cur = xh_nxt; xh_nxt = t;

        // e_h = new_e + e_h
        add_inplace_k<<<(int)(((size_t)E * H + 255) / 256), 256, 0, stream>>>(eh, tmpE2, (size_t)E * H);
    }

    // ---- decoders ----
    float* out = (float*)d_out;
    launch_gemm(stream, xh_cur, H, nullptr, nullptr, nullptr, nullptr, 0, dw_w1f, dw_b1, tmpN1, N, H, 1);
    dec_out_k<<<(int)(((size_t)N * 2 + 255) / 256), 256, 0, stream>>>(tmpN1, dw_w2, dw_b2, out, N, 2, 0);
    launch_gemm(stream, xh_cur, H, nullptr, nullptr, nullptr, nullptr, 0, dp_w1f, dp_b1, tmpN2, N, H, 1);
    dec_out_k<<<(int)(((size_t)N + 255) / 256), 256, 0, stream>>>(tmpN2, dp_w2, dp_b2, out, N, 1, 2);
#undef PF
}
